// EDGE_12438225289643
// MI455X (gfx1250) — compile-verified, matrix operand order checked by reading, not run on hardware
//
#include <hip/hip_runtime.h>
#include <hip/hip_bf16.h>
#include <cstdint>
#include <cstddef>

// ---------------------------------------------------------------------------
// MI455X / gfx1250 EdgeConv GNN pipeline.
// WMMA mapping: M = the 16 edges of one destination node, N = out channels,
// K = concat([x_i, x_j - x_i]) padded to multiples of 32 (f16, f32 accum).
// ---------------------------------------------------------------------------

typedef __attribute__((ext_vector_type(16))) _Float16 v16h;
typedef __attribute__((ext_vector_type(8)))  float    v8f;

#define WMMA16 __builtin_amdgcn_wmma_f32_16x16x32_f16

union Frag { unsigned u[8]; v16h h; };

static constexpr int NN   = 262144;   // nodes
static constexpr int DEG  = 16;
static constexpr int N2C  = 3600;     // clusters (16 graphs * 225)
static constexpr int BGR  = 16;       // graphs
static constexpr int NOUT = 101;

__device__ __forceinline__ void atomicMaxF32(float* addr, float val) {
  unsigned* ua = reinterpret_cast<unsigned*>(addr);
  unsigned cur = __float_as_uint(*addr);
  while (__uint_as_float(cur) < val) {
    unsigned assumed = cur;
    cur = atomicCAS(ua, assumed, __float_as_uint(val));
    if (cur == assumed) break;
  }
}

__device__ __forceinline__ float eluf(float v) {
  return (v > 0.0f) ? v : (expf(v) - 1.0f);
}

// ---------------------------------------------------------------------------
// Weight packing: f32 (Krows x Ncols) -> f16 B-fragment layout, one frag per
// (N-tile t, K-chunk kc). Element (lane, j): n = t*16 + lane%16,
// k0 = kc*32 + (j>=4?16:0) + (lane>=16?8:0) + (j&3)*2  (two f16 per dword).
// ---------------------------------------------------------------------------
__global__ void pack_weights_kernel(const float* __restrict__ w, int Krows, int Ncols,
                                    int T, int KC, unsigned* __restrict__ dst) {
  const int frag = blockIdx.x;             // t*KC + kc
  const int t = frag / KC, kc = frag % KC;
  const int tid = threadIdx.x;             // 0..255
  const int lane = tid >> 3, j = tid & 7;
  const int n    = t * 16 + (lane & 15);
  const int half = lane >> 4;
  const int k0   = kc * 32 + ((j >= 4) ? 16 : 0) + (half ? 8 : 0) + (j & 3) * 2;
  float v0 = (k0     < Krows && n < Ncols) ? w[(size_t)k0 * Ncols + n]       : 0.0f;
  float v1 = (k0 + 1 < Krows && n < Ncols) ? w[(size_t)(k0 + 1) * Ncols + n] : 0.0f;
  _Float16 h0 = (_Float16)v0, h1 = (_Float16)v1;
  unsigned u = ((unsigned)__builtin_bit_cast(unsigned short, h1) << 16) |
               (unsigned)__builtin_bit_cast(unsigned short, h0);
  dst[(size_t)frag * 256 + lane * 8 + j] = u;
}

// ---------------------------------------------------------------------------
// Core EdgeConv block. One wave handles one destination node per iteration:
//   16 edges -> feature tile in LDS -> WMMA GEMM1 -> relu -> LDS -> WMMA GEMM2
//   -> column max over edge rows -> +b2 -> (dense: ELU + write + BN stats)
//                                      (coarse: masked, atomic-max by cluster)
// ---------------------------------------------------------------------------
template <int CIN, int COUT, bool COARSE>
__global__ void __launch_bounds__(128)
edge_block_kernel(const _Float16* __restrict__ xin,      // [nIn, 32] padded f16
                  const int*      __restrict__ srcIdx,   // [N*16]
                  const int*      __restrict__ cluster,  // [N] (COARSE only)
                  const float*    __restrict__ b1,
                  const float*    __restrict__ b2,
                  const unsigned* __restrict__ w1p,
                  const unsigned* __restrict__ w2p,
                  float* __restrict__ out,               // dense: [N,COUT]; coarse: [N2C,32]
                  float* __restrict__ gsum,
                  float* __restrict__ gsq,
                  int nNodes) {
  constexpr int KC1 = (2 * CIN + 31) / 32;   // K chunks for GEMM1
  constexpr int T   = (COUT + 15) / 16;      // N tiles
  constexpr int NW  = 4;                     // waves / workgroup

  __shared__ _Float16 feat[NW][16][64];
  __shared__ _Float16 mid[NW][16][32];
  __shared__ float    rowNeg[NW][16];
  __shared__ float    sumS[32];
  __shared__ float    sqS[32];

  const int tid  = threadIdx.x;
  const int w    = tid >> 5;
  const int lane = tid & 31;
  const int k    = lane & 15;     // edge index within node / matrix row M
  const bool hi  = lane >= 16;

  if constexpr (!COARSE) {
    if (tid < 32) { sumS[tid] = 0.0f; sqS[tid] = 0.0f; }
  }
  __syncthreads();

  const int waveId     = blockIdx.x * NW + w;
  const int totalWaves = gridDim.x * NW;
  const int iters      = nNodes / totalWaves;   // launched so this is exact

  for (int it = 0; it < iters; ++it) {
    const int i = waveId + it * totalWaves;
    __syncthreads();   // feat/mid reuse across iterations

    // ---- gather: build 16 x (KC1*32) f16 feature tile ----
    int dstc = 0;
    {
      const int e = i * DEG + k;
      const int s = srcIdx[e];
      int gi, gj;
      if constexpr (COARSE) { gi = cluster[i]; gj = cluster[s]; dstc = gi; }
      else                  { gi = i;          gj = s; }
      const _Float16* xiRow = xin + (size_t)gi * 32;
      const _Float16* xjRow = xin + (size_t)gj * 32;
      _Float16* frow = &feat[w][k][0];
      if (!hi) {
#pragma unroll
        for (int c = 0; c < CIN; ++c) frow[c] = xiRow[c];
        if constexpr (COARSE) rowNeg[w][k] = (gi != gj) ? 0.0f : -__builtin_inff();
      } else {
#pragma unroll
        for (int c = 0; c < CIN; ++c)
          frow[CIN + c] = (_Float16)(xjRow[c] - xiRow[c]);
#pragma unroll
        for (int c = 2 * CIN; c < KC1 * 32; ++c) frow[c] = (_Float16)0.0f;
      }
    }
    __syncthreads();

    // ---- A fragments for GEMM1 (16-bit A layout, ISA 7.12.2) ----
    Frag a1[KC1];
#pragma unroll
    for (int kc = 0; kc < KC1; ++kc) {
#pragma unroll
      for (int j = 0; j < 8; ++j) {
        const int k0 = kc * 32 + ((j >= 4) ? 16 : 0) + (hi ? 8 : 0) + (j & 3) * 2;
        a1[kc].u[j] = *(const unsigned*)&feat[w][k][k0];
      }
    }

    // ---- GEMM1: relu(F @ W1 + b1) -> mid tile ----
#pragma unroll
    for (int t = 0; t < T; ++t) {
      const int c0 = t * 16 + k;
      const float bias = (c0 < COUT) ? b1[c0] : 0.0f;
      v8f acc = {bias, bias, bias, bias, bias, bias, bias, bias};
#pragma unroll
      for (int kc = 0; kc < KC1; ++kc) {
        Frag bfr;
        const unsigned* bp = w1p + (size_t)(t * KC1 + kc) * 256 + lane * 8;
#pragma unroll
        for (int j = 0; j < 8; ++j) bfr.u[j] = bp[j];
        acc = WMMA16(false, a1[kc].h, false, bfr.h, (short)0, acc, false, false);
      }
#pragma unroll
      for (int r = 0; r < 8; ++r) {
        const float v = fmaxf(acc[r], 0.0f);
        mid[w][r + (hi ? 8 : 0)][t * 16 + k] = (_Float16)v;
      }
    }
    __syncthreads();

    // ---- A fragment for GEMM2 (K = COUT, zero-pad chunks beyond COUT) ----
    Frag a2;
#pragma unroll
    for (int j = 0; j < 8; ++j) {
      const int k0 = ((j >= 4) ? 16 : 0) + (hi ? 8 : 0) + (j & 3) * 2;
      a2.u[j] = (k0 >= COUT) ? 0u : *(const unsigned*)&mid[w][k][k0];
    }

    // ---- GEMM2 + row-max epilogue ----
#pragma unroll
    for (int t = 0; t < T; ++t) {
      Frag bfr;
      const unsigned* bp = w2p + (size_t)t * 256 + lane * 8;
#pragma unroll
      for (int j = 0; j < 8; ++j) bfr.u[j] = bp[j];
      v8f acc = {0, 0, 0, 0, 0, 0, 0, 0};
      acc = WMMA16(false, a2.h, false, bfr.h, (short)0, acc, false, false);

      float cm = -__builtin_inff();
#pragma unroll
      for (int r = 0; r < 8; ++r) {
        float v = acc[r];
        if constexpr (COARSE) v += rowNeg[w][r + (hi ? 8 : 0)];  // -inf masks row
        cm = fmaxf(cm, v);
      }
      cm = fmaxf(cm, __shfl_xor(cm, 16, 32));   // combine M=0..7 with M=8..15
      const int c = t * 16 + k;
      if (c < COUT && !hi) {
        float val = cm + b2[c];
        if constexpr (COARSE) {
          atomicMaxF32(&out[(size_t)dstc * 32 + c], val);
        } else {
          val = eluf(val);
          out[(size_t)i * COUT + c] = val;
          atomicAdd(&sumS[c], val);
          atomicAdd(&sqS[c], val * val);
        }
      }
    }
  }

  if constexpr (!COARSE) {
    __syncthreads();
    if (tid < COUT) {
      atomicAdd(&gsum[tid], sumS[tid]);
      atomicAdd(&gsq[tid],  sqS[tid]);
    }
  }
}

// ---------------------------------------------------------------------------
// Small helper kernels
// ---------------------------------------------------------------------------
__global__ void fill_f32_kernel(float* p, float v, int n) {
  int i = blockIdx.x * blockDim.x + threadIdx.x;
  if (i < n) p[i] = v;
}

__global__ void convert_x_kernel(const float* __restrict__ x, _Float16* __restrict__ xf, int n) {
  int idx = blockIdx.x * blockDim.x + threadIdx.x;
  if (idx >= n * 32) return;
  int i = idx >> 5, c = idx & 31;
  xf[idx] = (c == 0) ? (_Float16)x[i] : (_Float16)0.0f;
}

__global__ void bn_params_kernel(const float* gsum, const float* gsq,
                                 const float* bnw, const float* bnb,
                                 float* scale, float* shift, int cout, float n) {
  int c = threadIdx.x;
  if (c < cout) {
    float mu  = gsum[c] / n;
    float var = gsq[c] / n - mu * mu;
    float s   = bnw[c] * rsqrtf(var + 1e-5f);
    scale[c] = s;
    shift[c] = bnb[c] - mu * s;
  }
}

template <int COUT, bool RES>
__global__ void bn_apply_kernel(const float* __restrict__ h,
                                const float* __restrict__ scale,
                                const float* __restrict__ shift,
                                const float* __restrict__ res,
                                float* __restrict__ yf32,
                                _Float16* __restrict__ yf16, int n) {
  int idx = blockIdx.x * blockDim.x + threadIdx.x;
  if (idx >= n * 32) return;
  int i = idx >> 5, c = idx & 31;
  if (c < COUT) {
    float v = h[(size_t)i * COUT + c] * scale[c] + shift[c];
    if constexpr (RES) v += res[(size_t)i * COUT + c];
    yf32[(size_t)i * COUT + c] = v;
    yf16[(size_t)i * 32 + c]   = (_Float16)v;
  } else {
    yf16[(size_t)i * 32 + c] = (_Float16)0.0f;
  }
}

// pool5: voxel cluster id, atomic-max features, count + position sums
__global__ void pool5_kernel(const float* __restrict__ y5, const float* __restrict__ pos,
                             int* __restrict__ cluster, float* __restrict__ x2max,
                             float* __restrict__ cnt, float* __restrict__ possum, int n) {
  int i = blockIdx.x * blockDim.x + threadIdx.x;
  if (i >= n) return;
  float px = pos[(size_t)i * 3 + 0];
  float py = pos[(size_t)i * 3 + 1];
  float pt = pos[(size_t)i * 3 + 2];
  int vx = min(max((int)(px / 16.0f), 0), 14);
  int vy = min(max((int)(py / 12.0f), 0), 14);
  int b  = i >> 14;                         // PG = 16384
  int cl = b * 225 + vy * 15 + vx;
  cluster[i] = cl;
  for (int c = 0; c < 32; ++c)
    atomicMaxF32(&x2max[(size_t)cl * 32 + c], y5[(size_t)i * 32 + c]);
  atomicAdd(&cnt[cl], 1.0f);
  atomicAdd(&possum[cl * 3 + 0], px);
  atomicAdd(&possum[cl * 3 + 1], py);
  atomicAdd(&possum[cl * 3 + 2], pt);
}

__global__ void pool5_fin_kernel(const float* __restrict__ x2max, const float* __restrict__ cnt,
                                 const float* __restrict__ possum, float* __restrict__ x2pool,
                                 _Float16* __restrict__ x2f16, float* __restrict__ pos2, int n2) {
  int idx = blockIdx.x * blockDim.x + threadIdx.x;
  if (idx >= n2 * 32) return;
  int i = idx >> 5, c = idx & 31;
  float v = x2max[idx];
  if (!__builtin_isfinite(v)) v = 0.0f;
  x2pool[idx] = v;
  x2f16[idx]  = (_Float16)v;
  if (c < 3) {
    float cc = cnt[i];
    pos2[i * 3 + c] = possum[i * 3 + c] / fmaxf(cc, 1.0f);
  }
}

// coarse finalize: where-finite -> ELU -> stats
__global__ void coarse_fin_kernel(const float* __restrict__ x2next, float* __restrict__ h2,
                                  float* __restrict__ gsum, float* __restrict__ gsq, int n2) {
  int idx = blockIdx.x * blockDim.x + threadIdx.x;
  if (idx >= n2 * 32) return;
  int c = idx & 31;
  float v = x2next[idx];
  if (!__builtin_isfinite(v)) v = 0.0f;
  v = eluf(v);
  h2[idx] = v;
  atomicAdd(&gsum[c], v);
  atomicAdd(&gsq[c], v * v);
}

// pool7 (4x4 voxel max into 16 slots/graph) + final FC [B,512]@[101,512]^T
__global__ void pool7_fc_kernel(const float* __restrict__ x2fin, const float* __restrict__ pos2,
                                const float* __restrict__ fcw, float* __restrict__ outp) {
  __shared__ float xp[BGR * 16 * 32];     // 32 KB
  const int tid = threadIdx.x;
  for (int idx = tid; idx < BGR * 16 * 32; idx += blockDim.x) xp[idx] = -__builtin_inff();
  __syncthreads();
  for (int j = tid; j < N2C; j += blockDim.x) {
    int b = j / 225;
    float px = pos2[j * 3 + 0], py = pos2[j * 3 + 1];
    int ux = min(max((int)(px / 60.0f), 0), 3);
    int uy = min(max((int)(py / 45.0f), 0), 3);
    int slot = uy * 4 + ux;
    float* dst = &xp[(b * 16 + slot) * 32];
    for (int c = 0; c < 32; ++c) atomicMaxF32(&dst[c], x2fin[(size_t)j * 32 + c]);
  }
  __syncthreads();
  for (int idx = tid; idx < BGR * NOUT; idx += blockDim.x) {
    int b = idx / NOUT, o = idx % NOUT;
    float acc = 0.0f;
    for (int s = 0; s < 512; ++s) {
      float v = xp[b * 512 + s];
      if (!__builtin_isfinite(v)) v = 0.0f;
      acc += v * fcw[(size_t)o * 512 + s];
    }
    outp[b * NOUT + o] = acc;
  }
}

// ---------------------------------------------------------------------------
// Host launcher
// ---------------------------------------------------------------------------
extern "C" void kernel_launch(void* const* d_in, const int* in_sizes, int n_in,
                              void* d_out, int out_size, void* d_ws, size_t ws_size,
                              hipStream_t stream) {
  (void)in_sizes; (void)n_in; (void)out_size; (void)ws_size;
  const float* x   = (const float*)d_in[0];
  const float* pos = (const float*)d_in[1];
  const int*   src = (const int*)d_in[2];              // edge_index row 0
  auto P = [&](int l, int j) { return (const float*)d_in[4 + (l - 1) * 6 + j]; };
  const float* fcw = (const float*)d_in[46];
  float* outp = (float*)d_out;

  // ---- workspace arena ----
  char* base = (char*)d_ws;
  size_t off = 0;
  auto alloc = [&](size_t bytes) -> void* {
    off = (off + 255) & ~(size_t)255;
    void* p = base + off;
    off += bytes;
    return p;
  };
  float*    hbuf  = (float*)alloc((size_t)NN * 32 * 4);
  float*    yA    = (float*)alloc((size_t)NN * 32 * 4);
  float*    ySC   = (float*)alloc((size_t)NN * 16 * 4);
  _Float16* f16A  = (_Float16*)alloc((size_t)NN * 32 * 2);
  _Float16* f16B  = (_Float16*)alloc((size_t)NN * 32 * 2);
  int*      clus  = (int*)alloc((size_t)NN * 4);
  float*    stats = (float*)alloc(64 * 4);          // gsum | gsq
  float*    bnp   = (float*)alloc(64 * 4);          // scale | shift
  float*    x2max = (float*)alloc((size_t)N2C * 32 * 4);
  float*    cnt   = (float*)alloc((size_t)N2C * 4);
  float*    psum  = (float*)alloc((size_t)N2C * 3 * 4);
  float*    pos2  = (float*)alloc((size_t)N2C * 3 * 4);
  float*    x2pl  = (float*)alloc((size_t)N2C * 32 * 4);
  _Float16* x2h16 = (_Float16*)alloc((size_t)N2C * 32 * 2);
  float*    x2nx  = (float*)alloc((size_t)N2C * 32 * 4);
  float*    h2    = (float*)alloc((size_t)N2C * 32 * 4);
  float*    y6f   = (float*)alloc((size_t)N2C * 32 * 4);
  _Float16* y6h   = (_Float16*)alloc((size_t)N2C * 32 * 2);
  float*    x2fin = (float*)alloc((size_t)N2C * 32 * 4);

  struct { int cin, cout; } L[8] = {{0,0},{1,8},{8,16},{16,16},{16,16},{16,32},{32,32},{32,32}};
  unsigned* w1p[8]; unsigned* w2p[8];
  for (int l = 1; l <= 7; ++l) {
    int T  = (L[l].cout + 15) / 16;
    int KC = (2 * L[l].cin + 31) / 32;
    w1p[l] = (unsigned*)alloc((size_t)T * KC * 256 * 4);
    w2p[l] = (unsigned*)alloc((size_t)T * 256 * 4);
    pack_weights_kernel<<<T * KC, 256, 0, stream>>>(P(l, 0), 2 * L[l].cin, L[l].cout, T, KC, w1p[l]);
    pack_weights_kernel<<<T, 256, 0, stream>>>(P(l, 2), L[l].cout, L[l].cout, T, 1, w2p[l]);
  }

  const dim3 EGRID(1024), EBLK(128);
  auto fill = [&](float* p, float v, int n) {
    fill_f32_kernel<<<(n + 255) / 256, 256, 0, stream>>>(p, v, n);
  };
  const int NT32 = (NN * 32 + 255) / 256;
  const int CT32 = (N2C * 32 + 255) / 256;

  convert_x_kernel<<<NT32, 256, 0, stream>>>(x, f16A, NN);

  // block 1: cin=1 -> cout=8
  fill(stats, 0.0f, 64);
  edge_block_kernel<1, 8, false><<<EGRID, EBLK, 0, stream>>>(
      f16A, src, nullptr, P(1,1), P(1,3), w1p[1], w2p[1], hbuf, stats, stats + 32, NN);
  bn_params_kernel<<<1, 32, 0, stream>>>(stats, stats + 32, P(1,4), P(1,5), bnp, bnp + 32, 8, (float)NN);
  bn_apply_kernel<8, false><<<NT32, 256, 0, stream>>>(hbuf, bnp, bnp + 32, nullptr, yA, f16B, NN);

  // block 2: 8 -> 16 (save post-BN output as skip)
  fill(stats, 0.0f, 64);
  edge_block_kernel<8, 16, false><<<EGRID, EBLK, 0, stream>>>(
      f16B, src, nullptr, P(2,1), P(2,3), w1p[2], w2p[2], hbuf, stats, stats + 32, NN);
  bn_params_kernel<<<1, 32, 0, stream>>>(stats, stats + 32, P(2,4), P(2,5), bnp, bnp + 32, 16, (float)NN);
  bn_apply_kernel<16, false><<<NT32, 256, 0, stream>>>(hbuf, bnp, bnp + 32, nullptr, ySC, f16A, NN);

  // block 3: 16 -> 16
  fill(stats, 0.0f, 64);
  edge_block_kernel<16, 16, false><<<EGRID, EBLK, 0, stream>>>(
      f16A, src, nullptr, P(3,1), P(3,3), w1p[3], w2p[3], hbuf, stats, stats + 32, NN);
  bn_params_kernel<<<1, 32, 0, stream>>>(stats, stats + 32, P(3,4), P(3,5), bnp, bnp + 32, 16, (float)NN);
  bn_apply_kernel<16, false><<<NT32, 256, 0, stream>>>(hbuf, bnp, bnp + 32, nullptr, yA, f16B, NN);

  // block 4: 16 -> 16, + skip(block2)
  fill(stats, 0.0f, 64);
  edge_block_kernel<16, 16, false><<<EGRID, EBLK, 0, stream>>>(
      f16B, src, nullptr, P(4,1), P(4,3), w1p[4], w2p[4], hbuf, stats, stats + 32, NN);
  bn_params_kernel<<<1, 32, 0, stream>>>(stats, stats + 32, P(4,4), P(4,5), bnp, bnp + 32, 16, (float)NN);
  bn_apply_kernel<16, true><<<NT32, 256, 0, stream>>>(hbuf, bnp, bnp + 32, ySC, yA, f16A, NN);

  // block 5: 16 -> 32
  fill(stats, 0.0f, 64);
  edge_block_kernel<16, 32, false><<<EGRID, EBLK, 0, stream>>>(
      f16A, src, nullptr, P(5,1), P(5,3), w1p[5], w2p[5], hbuf, stats, stats + 32, NN);
  bn_params_kernel<<<1, 32, 0, stream>>>(stats, stats + 32, P(5,4), P(5,5), bnp, bnp + 32, 32, (float)NN);
  bn_apply_kernel<32, false><<<NT32, 256, 0, stream>>>(hbuf, bnp, bnp + 32, nullptr, yA, f16B, NN);

  // pool5
  fill(x2max, -__builtin_inff(), N2C * 32);
  fill(cnt, 0.0f, N2C);
  fill(psum, 0.0f, N2C * 3);
  pool5_kernel<<<NN / 256, 256, 0, stream>>>(yA, pos, clus, x2max, cnt, psum, NN);
  pool5_fin_kernel<<<CT32, 256, 0, stream>>>(x2max, cnt, psum, x2pl, x2h16, pos2, N2C);

  // block 6 (coarse, masked): 32 -> 32
  fill(x2nx, -__builtin_inff(), N2C * 32);
  edge_block_kernel<32, 32, true><<<EGRID, EBLK, 0, stream>>>(
      x2h16, src, clus, P(6,1), P(6,3), w1p[6], w2p[6], x2nx, nullptr, nullptr, NN);
  fill(stats, 0.0f, 64);
  coarse_fin_kernel<<<CT32, 256, 0, stream>>>(x2nx, h2, stats, stats + 32, N2C);
  bn_params_kernel<<<1, 32, 0, stream>>>(stats, stats + 32, P(6,4), P(6,5), bnp, bnp + 32, 32, (float)N2C);
  bn_apply_kernel<32, false><<<CT32, 256, 0, stream>>>(h2, bnp, bnp + 32, nullptr, y6f, y6h, N2C);

  // block 7 (coarse, masked): 32 -> 32, + skip(pool5 output)
  fill(x2nx, -__builtin_inff(), N2C * 32);
  edge_block_kernel<32, 32, true><<<EGRID, EBLK, 0, stream>>>(
      y6h, src, clus, P(7,1), P(7,3), w1p[7], w2p[7], x2nx, nullptr, nullptr, NN);
  fill(stats, 0.0f, 64);
  coarse_fin_kernel<<<CT32, 256, 0, stream>>>(x2nx, h2, stats, stats + 32, N2C);
  bn_params_kernel<<<1, 32, 0, stream>>>(stats, stats + 32, P(7,4), P(7,5), bnp, bnp + 32, 32, (float)N2C);
  bn_apply_kernel<32, true><<<CT32, 256, 0, stream>>>(h2, bnp, bnp + 32, x2pl, x2fin, y6h, N2C);

  // pool7 + FC
  pool7_fc_kernel<<<1, 256, 0, stream>>>(x2fin, pos2, fcw, outp);
}
